// ODE_RNN_4827543241512
// MI455X (gfx1250) — compile-verified, matrix operand order checked by reading, not run on hardware
//
#include <hip/hip_runtime.h>
#include <hip/hip_bf16.h>
#include <math.h>

// ---------------- problem constants ----------------
#define K_H     512
#define K_OUT   32
#define KX      128              // K_EXPAND + K_IN
#define STATE_W (K_OUT + K_H)    // 544
#define T_END   0.1f

// ---------------- tiling ----------------
#define ROWS     32              // batch rows per workgroup (2 WMMA M-tiles)
#define NTHREADS 256             // 8 wave32s
#define NWAVES   8

// LDS strides (elements), all chosen so every 8-short chunk is 16B aligned
#define ACC_S 516                // f32 rows
#define H0_S  520                // bf16 rows
#define CUR_S 520
#define Z_S   1032
#define X_S   136

typedef __attribute__((ext_vector_type(16))) __bf16 v16bf;
typedef __attribute__((ext_vector_type(8)))  float  v8f;

__device__ __forceinline__ unsigned short f2bf(float f) {
  unsigned int u = __float_as_uint(f);
  u = u + 0x7FFFu + ((u >> 16) & 1u);        // round-to-nearest-even
  return (unsigned short)(u >> 16);
}
__device__ __forceinline__ float bf2f(unsigned short h) {
  return __uint_as_float(((unsigned int)h) << 16);
}
__device__ __forceinline__ float sigmoidf(float x) {
  return 1.0f / (1.0f + __expf(-x));
}
__device__ __forceinline__ v8f splat8(float v) {
  v8f r; for (int j = 0; j < 8; ++j) r[j] = v; return r;
}

union FragU { v16bf v; uint4 q[2]; };

// A-matrix (16x32 bf16) fragment from row-major LDS tile.
// ISA layout: lane<16 holds M=lane, K={k0..k0+7, k0+16..k0+23};
//             lane>=16 holds M=lane-16, K={k0+8..k0+15, k0+24..k0+31}.
__device__ __forceinline__ v16bf load_frag_lds(const unsigned short* base, int stride,
                                               int r0, int k0, int lane) {
  int r = r0 + (lane & 15);
  int k = k0 + ((lane >> 4) << 3);
  const unsigned short* p = base + r * stride + k;
  FragU f;
  f.q[0] = *(const uint4*)(p);
  f.q[1] = *(const uint4*)(p + 16);
  return f.v;
}
// B-matrix (32x16 bf16) fragment from row-major (N,K) bf16 weights in global
// (mirror of the A layout with N in place of M).
__device__ __forceinline__ v16bf load_frag_glb(const unsigned short* __restrict__ W, int ldk,
                                               int n0, int k0, int lane) {
  int n = n0 + (lane & 15);
  int k = k0 + ((lane >> 4) << 3);
  const unsigned short* p = W + (size_t)n * ldk + k;
  FragU f;
  f.q[0] = *(const uint4*)(p);
  f.q[1] = *(const uint4*)(p + 16);
  return f.v;
}

#define WMMA_BF16(a,b,c) \
  __builtin_amdgcn_wmma_f32_16x16x32_bf16(false,(a),false,(b),(short)0,(c),false,false)

// ---------------- weight f32 -> bf16 pre-pass (into d_ws) ----------------
__global__ void prep_weights(const float* __restrict__ W1, const float* __restrict__ W2,
                             const float* __restrict__ Wih, const float* __restrict__ Whh,
                             const float* __restrict__ Wy, unsigned short* __restrict__ ws) {
  const int n1 = 2 * K_H * K_H;        // W1  (1024x512)
  const int n2 = n1 + K_H * 2 * K_H;   // W2  (512x1024)
  const int n3 = n2 + 3 * K_H * KX;    // Wih (1536x128)
  const int n4 = n3 + 3 * K_H * K_H;   // Whh (1536x512)
  const int n5 = n4 + K_OUT * K_H;     // Wy  (32x512)
  for (int i = blockIdx.x * blockDim.x + threadIdx.x; i < n5; i += gridDim.x * blockDim.x) {
    float v;
    if      (i < n1) v = W1[i];
    else if (i < n2) v = W2[i - n1];
    else if (i < n3) v = Wih[i - n2];
    else if (i < n4) v = Whh[i - n3];
    else             v = Wy[i - n4];
    ws[i] = f2bf(v);
  }
}

// ---------------- fused ODE-RNN step ----------------
__global__ __launch_bounds__(NTHREADS)
void ode_rnn_step(const float* __restrict__ state, const float* __restrict__ xt,
                  const float* __restrict__ b1, const float* __restrict__ b2,
                  const float* __restrict__ b_ih, const float* __restrict__ b_hh,
                  const float* __restrict__ by, const unsigned char* __restrict__ y_mask,
                  const unsigned short* __restrict__ wsb, float* __restrict__ out)
{
  extern __shared__ __align__(16) char smem[];
  float*          accf = (float*)smem;                                // [ROWS][ACC_S] RK4 acc / h
  unsigned short* h0b  = (unsigned short*)(smem + ROWS * ACC_S * 4);  // [ROWS][H0_S]  bf16(h0)
  unsigned short* curb = h0b + ROWS * H0_S;                           // [ROWS][CUR_S] bf16 A-src
  unsigned short* zb   = curb + ROWS * CUR_S;                         // [ROWS][Z_S]   tanh layer / h_new
  unsigned short* xb   = zb + ROWS * Z_S;                             // [ROWS][X_S]   bf16(xt)

  const unsigned short* W1b  = wsb;
  const unsigned short* W2b  = W1b + 2 * K_H * K_H;
  const unsigned short* Wihb = W2b + K_H * 2 * K_H;
  const unsigned short* Whhb = Wihb + 3 * K_H * KX;
  const unsigned short* Wyb  = Whhb + 3 * K_H * K_H;

  const int tid  = threadIdx.x;
  const int lane = tid & 31;
  const int wave = tid >> 5;
  const int r0   = blockIdx.x * ROWS;

  // ---- load state tile ----
  for (int i = tid; i < ROWS * K_H; i += NTHREADS) {
    int r = i >> 9, c = i & (K_H - 1);
    float h = state[(size_t)(r0 + r) * STATE_W + K_OUT + c];
    accf[r * ACC_S + c] = h;
    unsigned short hb = f2bf(h);
    h0b[r * H0_S + c]  = hb;
    curb[r * CUR_S + c] = hb;
  }
  for (int i = tid; i < ROWS * KX; i += NTHREADS) {
    int r = i >> 7, c = i & (KX - 1);
    xb[r * X_S + c] = f2bf(xt[(size_t)(r0 + r) * KX + c]);
  }
  __syncthreads();

  const float kw[4] = {T_END / 6.f, T_END / 3.f, T_END / 3.f, T_END / 6.f};
  const float cn[4] = {0.5f * T_END, 0.5f * T_END, T_END, 0.f};

  const int lmsk = lane & 15;
  const int mhi  = (lane >> 4) << 3;   // C-fragment row base offset per lane half

  // ---- RK4: 4 evals of f(y) = tanh(y W1^T + b1) W2^T + b2 ----
  for (int e = 0; e < 4; ++e) {
    // stage 1: zb = tanh(cur @ W1^T + b1), N = 1024.  2M x 4N register blocking.
    for (int job = wave; job < (2 * K_H) / 64; job += NWAVES) {   // 16 jobs
      int n0 = job * 64;
      v8f acc[2][4];
#pragma unroll
      for (int ni = 0; ni < 4; ++ni) {
        float bv = b1[n0 + ni * 16 + lmsk];
        acc[0][ni] = splat8(bv);
        acc[1][ni] = splat8(bv);
      }
      for (int k0 = 0; k0 < K_H; k0 += 32) {
        v16bf am0 = load_frag_lds(curb, CUR_S, 0,  k0, lane);
        v16bf am1 = load_frag_lds(curb, CUR_S, 16, k0, lane);
#pragma unroll
        for (int ni = 0; ni < 4; ++ni) {
          v16bf b = load_frag_glb(W1b, K_H, n0 + ni * 16, k0, lane);
          acc[0][ni] = WMMA_BF16(am0, b, acc[0][ni]);
          acc[1][ni] = WMMA_BF16(am1, b, acc[1][ni]);
        }
      }
#pragma unroll
      for (int mi = 0; mi < 2; ++mi)
#pragma unroll
        for (int ni = 0; ni < 4; ++ni) {
          int nn = n0 + ni * 16 + lmsk;
          for (int j = 0; j < 8; ++j)
            zb[(mi * 16 + mhi + j) * Z_S + nn] = f2bf(tanhf(acc[mi][ni][j]));
        }
    }
    __syncthreads();
    // stage 2: k = zb @ W2^T + b2; acc += w_e*k; cur = bf16(h0 + c_e*k).  2M x 4N.
    for (int job = wave; job < K_H / 64; job += NWAVES) {         // 8 jobs
      int n0 = job * 64;
      v8f acc[2][4];
#pragma unroll
      for (int ni = 0; ni < 4; ++ni) {
        float bv = b2[n0 + ni * 16 + lmsk];
        acc[0][ni] = splat8(bv);
        acc[1][ni] = splat8(bv);
      }
      for (int k0 = 0; k0 < 2 * K_H; k0 += 32) {
        v16bf am0 = load_frag_lds(zb, Z_S, 0,  k0, lane);
        v16bf am1 = load_frag_lds(zb, Z_S, 16, k0, lane);
#pragma unroll
        for (int ni = 0; ni < 4; ++ni) {
          v16bf b = load_frag_glb(W2b, 2 * K_H, n0 + ni * 16, k0, lane);
          acc[0][ni] = WMMA_BF16(am0, b, acc[0][ni]);
          acc[1][ni] = WMMA_BF16(am1, b, acc[1][ni]);
        }
      }
#pragma unroll
      for (int mi = 0; mi < 2; ++mi)
#pragma unroll
        for (int ni = 0; ni < 4; ++ni) {
          int nn = n0 + ni * 16 + lmsk;
          for (int j = 0; j < 8; ++j) {
            int m = mi * 16 + mhi + j;
            float kv = acc[mi][ni][j];
            float a2 = accf[m * ACC_S + nn] + kw[e] * kv;
            accf[m * ACC_S + nn] = a2;
            if (e < 3) curb[m * CUR_S + nn] = f2bf(bf2f(h0b[m * H0_S + nn]) + cn[e] * kv);
            else       curb[m * CUR_S + nn] = f2bf(a2);   // cur = bf16(h_) for GRU
          }
        }
    }
    __syncthreads();
  }

  // ---- GRU cell: h = (1-z)*tanh(i_n + r*h_n) + z*h_.  2M x 1N, 6 gates share A. ----
  for (int job = wave; job < K_H / 16; job += NWAVES) {           // 32 jobs
    int n0 = job * 16;
    int nl = n0 + lmsk;
    v8f ir0 = splat8(b_ih[nl]),           ir1 = ir0;
    v8f iz0 = splat8(b_ih[K_H + nl]),     iz1 = iz0;
    v8f in0 = splat8(b_ih[2 * K_H + nl]), in1 = in0;
    v8f hr0 = splat8(b_hh[nl]),           hr1 = hr0;
    v8f hz0 = splat8(b_hh[K_H + nl]),     hz1 = hz0;
    v8f hn0 = splat8(b_hh[2 * K_H + nl]), hn1 = hn0;
    for (int k0 = 0; k0 < KX; k0 += 32) {
      v16bf am0 = load_frag_lds(xb, X_S, 0,  k0, lane);
      v16bf am1 = load_frag_lds(xb, X_S, 16, k0, lane);
      v16bf b;
      b = load_frag_glb(Wihb, KX, n0,           k0, lane);
      ir0 = WMMA_BF16(am0, b, ir0); ir1 = WMMA_BF16(am1, b, ir1);
      b = load_frag_glb(Wihb, KX, K_H + n0,     k0, lane);
      iz0 = WMMA_BF16(am0, b, iz0); iz1 = WMMA_BF16(am1, b, iz1);
      b = load_frag_glb(Wihb, KX, 2 * K_H + n0, k0, lane);
      in0 = WMMA_BF16(am0, b, in0); in1 = WMMA_BF16(am1, b, in1);
    }
    for (int k0 = 0; k0 < K_H; k0 += 32) {
      v16bf am0 = load_frag_lds(curb, CUR_S, 0,  k0, lane);
      v16bf am1 = load_frag_lds(curb, CUR_S, 16, k0, lane);
      v16bf b;
      b = load_frag_glb(Whhb, K_H, n0,           k0, lane);
      hr0 = WMMA_BF16(am0, b, hr0); hr1 = WMMA_BF16(am1, b, hr1);
      b = load_frag_glb(Whhb, K_H, K_H + n0,     k0, lane);
      hz0 = WMMA_BF16(am0, b, hz0); hz1 = WMMA_BF16(am1, b, hz1);
      b = load_frag_glb(Whhb, K_H, 2 * K_H + n0, k0, lane);
      hn0 = WMMA_BF16(am0, b, hn0); hn1 = WMMA_BF16(am1, b, hn1);
    }
#pragma unroll
    for (int mi = 0; mi < 2; ++mi) {
      v8f ir = mi ? ir1 : ir0, iz = mi ? iz1 : iz0, in_ = mi ? in1 : in0;
      v8f hr = mi ? hr1 : hr0, hz = mi ? hz1 : hz0, hn  = mi ? hn1 : hn0;
      int nn = n0 + lmsk;
      for (int j = 0; j < 8; ++j) {
        int m = mi * 16 + mhi + j;
        float r = sigmoidf(ir[j] + hr[j]);
        float z = sigmoidf(iz[j] + hz[j]);
        float n = tanhf(in_[j] + r * hn[j]);
        float hprev = accf[m * ACC_S + nn];
        float hnew  = (1.f - z) * n + z * hprev;
        accf[m * ACC_S + nn] = hnew;                 // column-exclusive, safe in place
        zb[m * Z_S + nn] = f2bf(hnew);               // reuse zb as bf16(h_new) for the head
        out[(size_t)(r0 + m) * STATE_W + K_OUT + nn] = hnew;
      }
    }
  }
  __syncthreads();

  // ---- output head: ly = h_new @ Wy^T + by ; nyt = mask ? ly : yt ----
  for (int job = wave; job < (K_OUT / 16) * 2; job += NWAVES) {   // 4 jobs
    int n0 = (job >> 1) * 16, m0 = (job & 1) * 16;
    v8f acc = splat8(by[n0 + lmsk]);
    for (int k0 = 0; k0 < K_H; k0 += 32) {
      v16bf a = load_frag_lds(zb, Z_S, m0, k0, lane);
      acc = WMMA_BF16(a, load_frag_glb(Wyb, K_H, n0, k0, lane), acc);
    }
    int nn = n0 + lmsk;
    for (int j = 0; j < 8; ++j) {
      int m = m0 + mhi + j;
      float yprev = state[(size_t)(r0 + m) * STATE_W + nn];
      out[(size_t)(r0 + m) * STATE_W + nn] = y_mask[nn] ? acc[j] : yprev;
    }
  }
}

extern "C" void kernel_launch(void* const* d_in, const int* in_sizes, int n_in,
                              void* d_out, int out_size, void* d_ws, size_t ws_size,
                              hipStream_t stream) {
  const float* state = (const float*)d_in[0];
  const float* xt    = (const float*)d_in[1];
  // d_in[2]=dt, d_in[3]=ti are unused by the reference (it uses T_END)
  const float* W1  = (const float*)d_in[4];
  const float* b1  = (const float*)d_in[5];
  const float* W2  = (const float*)d_in[6];
  const float* b2  = (const float*)d_in[7];
  const float* Wih = (const float*)d_in[8];
  const float* Whh = (const float*)d_in[9];
  const float* bih = (const float*)d_in[10];
  const float* bhh = (const float*)d_in[11];
  const float* Wy  = (const float*)d_in[12];
  const float* by  = (const float*)d_in[13];
  const unsigned char* y_mask = (const unsigned char*)d_in[14];
  float* out = (float*)d_out;
  unsigned short* wsb = (unsigned short*)d_ws;

  const int batch = in_sizes[0] / STATE_W;

  prep_weights<<<512, 256, 0, stream>>>(W1, W2, Wih, Whh, Wy, wsb);

  const size_t lds = (size_t)ROWS * ACC_S * 4 +
                     (size_t)ROWS * (H0_S + CUR_S) * 2 +
                     (size_t)ROWS * Z_S * 2 +
                     (size_t)ROWS * X_S * 2;   // ~203 KB of CDNA5's 320 KB/WGP

  ode_rnn_step<<<batch / ROWS, NTHREADS, lds, stream>>>(
      state, xt, b1, b2, bih, bhh, by, y_mask, wsb, out);
}
